// Pointnet_ssg_84293028152071
// MI455X (gfx1250) — matrix operand order repaired and checked by reading, not compiled, not run on hardware
//
#include <hip/hip_runtime.h>
#include <hip/hip_bf16.h>
#include <stdint.h>
#include <stddef.h>

#define EPSBN 1e-5f

typedef __bf16 bf16;
typedef __attribute__((ext_vector_type(16))) __bf16 bf16x16;
typedef __attribute__((ext_vector_type(8)))  float  floatx8;

// ---------------------------------------------------------------------------
// 0. Split (B,6,N) xyz into coords (B,N,3) and feats (B,N,3)
// ---------------------------------------------------------------------------
__global__ void split_xyz(const float* __restrict__ xyz, float* __restrict__ coords,
                          float* __restrict__ feats, int B, int N) {
  int i = blockIdx.x * blockDim.x + threadIdx.x;
  if (i >= B * N) return;
  int b = i / N, n = i % N;
  const float* src = xyz + (size_t)b * 6 * N + n;
  float* cd = coords + (size_t)i * 3;
  float* ft = feats + (size_t)i * 3;
  cd[0] = src[0];
  cd[1] = src[(size_t)N];
  cd[2] = src[2 * (size_t)N];
  ft[0] = src[3 * (size_t)N];
  ft[1] = src[4 * (size_t)N];
  ft[2] = src[5 * (size_t)N];
}

// ---------------------------------------------------------------------------
// 1. Farthest point sampling: one block per batch, LDS-resident distances,
//    deterministic argmax (ties -> lowest index, matches jnp.argmax).
// ---------------------------------------------------------------------------
__global__ __launch_bounds__(256) void fps_kernel(const float* __restrict__ coords,
                                                  int* __restrict__ fidx,
                                                  int N, int npoint) {
  __shared__ float dist[4096];
  __shared__ float rv[256];
  __shared__ int   ri[256];
  __shared__ float ctr[3];
  int b = blockIdx.x;
  const float* base = coords + (size_t)b * N * 3;
  for (int n = threadIdx.x; n < N; n += 256) dist[n] = 1e10f;
  int far = 0;
  for (int i = 0; i < npoint; ++i) {
    if (threadIdx.x == 0) {
      fidx[b * npoint + i] = far;
      ctr[0] = base[far * 3 + 0];
      ctr[1] = base[far * 3 + 1];
      ctr[2] = base[far * 3 + 2];
    }
    __syncthreads();
    float cx = ctr[0], cy = ctr[1], cz = ctr[2];
    float bv = -1.0f; int bi = 0;
    for (int n = threadIdx.x; n < N; n += 256) {
      float dx = base[n * 3 + 0] - cx;
      float dy = base[n * 3 + 1] - cy;
      float dz = base[n * 3 + 2] - cz;
      float d = dx * dx + dy * dy + dz * dz;
      float dd = fminf(dist[n], d);
      dist[n] = dd;
      if (dd > bv) { bv = dd; bi = n; }  // ascending scan keeps lowest idx on tie
    }
    rv[threadIdx.x] = bv; ri[threadIdx.x] = bi;
    __syncthreads();
    for (int off = 128; off > 0; off >>= 1) {
      if (threadIdx.x < off) {
        float v2 = rv[threadIdx.x + off]; int i2 = ri[threadIdx.x + off];
        if (v2 > rv[threadIdx.x] || (v2 == rv[threadIdx.x] && i2 < ri[threadIdx.x])) {
          rv[threadIdx.x] = v2; ri[threadIdx.x] = i2;
        }
      }
      __syncthreads();
    }
    far = ri[0];
    __syncthreads();
  }
}

// ---------------------------------------------------------------------------
// 2. Ball query: ascending-index scan collects first K points within radius
//    (== reference's sort-then-truncate); pad with first hit.
// ---------------------------------------------------------------------------
__global__ void ball_group(const float* __restrict__ coords, const int* __restrict__ fidx,
                           float* __restrict__ newc, int* __restrict__ gidx,
                           int B, int N, int S, int K, float r2) {
  int i = blockIdx.x * blockDim.x + threadIdx.x;
  if (i >= B * S) return;
  int b = i / S;
  int f = fidx[i];
  const float* base = coords + (size_t)b * N * 3;
  float nx = base[f * 3 + 0], ny = base[f * 3 + 1], nz = base[f * 3 + 2];
  newc[(size_t)i * 3 + 0] = nx;
  newc[(size_t)i * 3 + 1] = ny;
  newc[(size_t)i * 3 + 2] = nz;
  int* gp = gidx + (size_t)i * K;
  int cnt = 0, first = -1;
  for (int n = 0; n < N && cnt < K; ++n) {
    float dx = base[n * 3 + 0] - nx;
    float dy = base[n * 3 + 1] - ny;
    float dz = base[n * 3 + 2] - nz;
    if (dx * dx + dy * dy + dz * dz <= r2) {
      if (first < 0) first = n;
      gp[cnt++] = n;
    }
  }
  for (int k = cnt; k < K; ++k) gp[k] = first;  // centroid itself guarantees first>=0
}

// ---------------------------------------------------------------------------
// 3. Build grouped activation matrix A (M = B*S*K rows, Kpad cols, bf16),
//    rows = [coords-diff(3), feats(Cf), zero-pad].
// ---------------------------------------------------------------------------
__global__ void build_grouped(const float* __restrict__ coords, const float* __restrict__ feats,
                              const int* __restrict__ gidx, const float* __restrict__ newc,
                              bf16* __restrict__ A, int N, int S, int K, int Cf, int Kpad,
                              size_t total) {
  size_t i = (size_t)blockIdx.x * blockDim.x + threadIdx.x;
  if (i >= total) return;
  int n = gidx[i];
  size_t bs = i / K;
  int b = (int)(bs / S);
  bf16* row = A + i * Kpad;
  const float* c  = coords + ((size_t)b * N + n) * 3;
  const float* nc = newc + bs * 3;
  row[0] = (bf16)(c[0] - nc[0]);
  row[1] = (bf16)(c[1] - nc[1]);
  row[2] = (bf16)(c[2] - nc[2]);
  const float* fp = feats + ((size_t)b * N + n) * Cf;
  for (int cc = 0; cc < Cf; ++cc) row[3 + cc] = (bf16)fp[cc];
  for (int cc = 3 + Cf; cc < Kpad; ++cc) row[cc] = (bf16)0.0f;
}

// group_all variant: rows = [raw coords(3), feats(Cf), pad]; M = B*Np
__global__ void build_group_all(const float* __restrict__ coords, const float* __restrict__ feats,
                                bf16* __restrict__ A, int total, int Cf, int Kpad) {
  int i = blockIdx.x * blockDim.x + threadIdx.x;
  if (i >= total) return;
  bf16* row = A + (size_t)i * Kpad;
  const float* c = coords + (size_t)i * 3;
  row[0] = (bf16)c[0]; row[1] = (bf16)c[1]; row[2] = (bf16)c[2];
  const float* fp = feats + (size_t)i * Cf;
  for (int cc = 0; cc < Cf; ++cc) row[3 + cc] = (bf16)fp[cc];
  for (int cc = 3 + Cf; cc < Kpad; ++cc) row[cc] = (bf16)0.0f;
}

// ---------------------------------------------------------------------------
// 4. Weight pack into WMMA B-fragment blocked layout:
//    WPB[(kb*(N/16)+nb)*512 + lane*16 + e] = W[nb*16+(lane&15)][kb*32+kk]  (0-pad K)
//    kk = e<8 ? 8*(lane>>4)+e : 16+8*(lane>>4)+(e-8); lane&15 = N column within tile.
//    Each lane's fragment is then a contiguous 32B run in memory.
// ---------------------------------------------------------------------------
__global__ void pack_w_blocked(const float* __restrict__ Wsrc, bf16* __restrict__ WPB,
                               int Cin, int N, size_t total) {
  size_t i = (size_t)blockIdx.x * blockDim.x + threadIdx.x;
  if (i >= total) return;
  int e    = (int)(i & 15);
  int lane = (int)((i >> 4) & 31);
  size_t blk = i >> 9;
  int NT = N >> 4;
  int nb = (int)(blk % NT);
  int kb = (int)(blk / NT);
  int n = nb * 16 + (lane & 15);
  int k = kb * 32 + ((e < 8) ? ((lane >> 4) * 8 + e) : (16 + (lane >> 4) * 8 + (e - 8)));
  WPB[i] = (k < Cin) ? (bf16)Wsrc[(size_t)n * Cin + k] : (bf16)0.0f;
}

// ---------------------------------------------------------------------------
// 5. WMMA GEMM: Y[M,N] = A[M,K](bf16) * W[N,K]^T(bf16) + bias, f32 accumulate.
//    No LDS, no barriers: B fragments are pre-blocked in memory (contiguous
//    1KB per wave per tile, L2-resident and shared across blocks); A fragments
//    are two contiguous 16B runs per lane (-> global_load_b128 pairs).
//    Each wave computes a 16x32 tile (one A fragment feeds two WMMAs).
// ---------------------------------------------------------------------------
__global__ __launch_bounds__(256) void gemm_bf16_wmma(
    const bf16* __restrict__ A, const bf16* __restrict__ WPB,
    const float* __restrict__ bias, float* __restrict__ Y,
    int M, int N, int K) {
  const int wave = threadIdx.x >> 5;
  const int lane = threadIdx.x & 31;
  const int NT = N >> 4;                       // 16-wide n-tiles
  const int nb = blockIdx.x * 2;               // this block's first n-tile
  const int mBase = ((int)blockIdx.y * 8 + wave) * 16;
  const bool act = (mBase < M);
  const int mrow = act ? (mBase + (lane & 15)) : (lane & 15);  // M>=16 always
  const int half = lane >> 4;
  const unsigned short* arow = (const unsigned short*)(A + (size_t)mrow * K);
  const bf16* bbase = WPB + (size_t)lane * 16 + (size_t)nb * 512;

  floatx8 acc0 = {}, acc1 = {};
  union FragA { bf16x16 v; unsigned int u[8]; } fa;

  const int KB = K >> 5;
#pragma unroll 2
  for (int kb = 0; kb < KB; ++kb) {
    const int kc = kb << 5;
    __builtin_prefetch((const void*)(arow + kc + 64), 0, 1);
#pragma unroll
    for (int vg = 0; vg < 8; ++vg) {
      int kbb = (vg < 4) ? (vg * 2) : (16 + (vg - 4) * 2);
      fa.u[vg] = *(const unsigned int*)(arow + kc + kbb + half * 8);
    }
    const bf16* bp = bbase + (size_t)kb * NT * 512;
    bf16x16 fb0 = *(const bf16x16*)(bp);
    bf16x16 fb1 = *(const bf16x16*)(bp + 512);
    acc0 = __builtin_amdgcn_wmma_f32_16x16x32_bf16(false, fa.v, false, fb0,
                                                   (short)0, acc0, false, false);
    acc1 = __builtin_amdgcn_wmma_f32_16x16x32_bf16(false, fa.v, false, fb1,
                                                   (short)0, acc1, false, false);
  }

  if (act) {
    int col0 = nb * 16 + (lane & 15);
    int col1 = col0 + 16;
    float bv0 = bias[col0], bv1 = bias[col1];
#pragma unroll
    for (int r = 0; r < 8; ++r) {
      int mm = mBase + ((lane >> 4) ? (8 + r) : r);
      Y[(size_t)mm * N + col0] = acc0[r] + bv0;
      Y[(size_t)mm * N + col1] = acc1[r] + bv1;
    }
  }
}

// ---------------------------------------------------------------------------
// 6. BatchNorm stats, deterministic two-stage, coalesced:
//    stage 1: block = (64-channel group) x (row block); lanes read 64
//    consecutive channels per row; fixed-order LDS combine of the 4 strips.
//    stage 2: fixed-order sum of row-block partials -> mean/var (ddof=0).
// ---------------------------------------------------------------------------
__global__ __launch_bounds__(256) void bn_stats_partial(
    const float* __restrict__ Y, int M, int N, int RB,
    float* __restrict__ pS, float* __restrict__ pQ) {
  __shared__ float s1[256], s2[256];
  int d = blockIdx.x * 64 + (threadIdx.x & 63);
  int rb = blockIdx.y;
  int strip = threadIdx.x >> 6;
  int mChunk = (M + RB - 1) / RB;
  int m0 = rb * mChunk;
  int m1 = m0 + mChunk; if (m1 > M) m1 = M;
  float a = 0.0f, b = 0.0f;
  for (int m = m0 + strip; m < m1; m += 4) {
    float v = Y[(size_t)m * N + d];
    a += v; b += v * v;
  }
  s1[threadIdx.x] = a; s2[threadIdx.x] = b;
  __syncthreads();
  if (threadIdx.x < 64) {
    int t = threadIdx.x;
    a = ((s1[t] + s1[t + 64]) + (s1[t + 128] + s1[t + 192]));
    b = ((s2[t] + s2[t + 64]) + (s2[t + 128] + s2[t + 192]));
    pS[(size_t)rb * N + d] = a;
    pQ[(size_t)rb * N + d] = b;
  }
}

__global__ void bn_stats_final(const float* __restrict__ pS, const float* __restrict__ pQ,
                               int N, int RB, int M,
                               float* __restrict__ mean, float* __restrict__ var) {
  int d = blockIdx.x * blockDim.x + threadIdx.x;
  if (d >= N) return;
  float a = 0.0f, b = 0.0f;
  for (int rb = 0; rb < RB; ++rb) {
    a += pS[(size_t)rb * N + d];
    b += pQ[(size_t)rb * N + d];
  }
  float mu = a / (float)M;
  mean[d] = mu;
  var[d]  = b / (float)M - mu * mu;
}

// 7. Fused BN + ReLU: f32 in-place (for max-pool) + bf16 copy (next GEMM A).
__global__ void bn_relu_kernel(float* __restrict__ Y, bf16* __restrict__ Abf,
                               const float* __restrict__ mean, const float* __restrict__ var,
                               const float* __restrict__ g, const float* __restrict__ be,
                               size_t total, int N) {
  size_t i = (size_t)blockIdx.x * blockDim.x + threadIdx.x;
  if (i >= total) return;
  int d = (int)(i % (size_t)N);
  float x = Y[i];
  x = (x - mean[d]) * rsqrtf(var[d] + EPSBN) * g[d] + be[d];
  x = fmaxf(x, 0.0f);
  Y[i]   = x;
  Abf[i] = (bf16)x;
}

// 8. Max-pool over the K neighbor axis: (BS*K, D) -> (BS, D).
__global__ void max_over_k(const float* __restrict__ Y, float* __restrict__ out,
                           int BS, int K, int D) {
  int i = blockIdx.x * blockDim.x + threadIdx.x;
  if (i >= BS * D) return;
  int bs = i / D, d = i % D;
  const float* p = Y + ((size_t)bs * K) * D + d;
  float m = p[0];
  for (int k = 1; k < K; ++k) m = fmaxf(m, p[(size_t)k * D]);
  out[i] = m;
}

// 9. f32 -> bf16 convert (FC input).
__global__ void cvt_bf16(const float* __restrict__ x, bf16* __restrict__ y, int n) {
  int i = blockIdx.x * blockDim.x + threadIdx.x;
  if (i < n) y[i] = (bf16)x[i];
}

// ---------------------------------------------------------------------------
// Host orchestration
// ---------------------------------------------------------------------------
static inline size_t pad32(size_t x) { return (x + 31) & ~(size_t)31; }

extern "C" void kernel_launch(void* const* d_in, const int* in_sizes, int n_in,
                              void* d_out, int out_size, void* d_ws, size_t ws_size,
                              hipStream_t stream) {
  (void)in_sizes; (void)n_in; (void)out_size; (void)ws_size;
  const int B = 16, N0 = 4096;
  static const int   npointA[4] = {512, 256, 128, 32};
  static const int   nsampA[4]  = {32, 128, 64, 128};
  static const float radA[4]    = {0.2f, 0.4f, 0.6f, 0.8f};
  static const int   mlps[5][3] = {{64,64,128},{128,128,256},{256,256,512},
                                   {512,512,1024},{1024,1024,2048}};

  auto saP = [&](int st, int l, int j) { return (const float*)d_in[1 + st * 12 + l * 4 + j]; };
  auto fcP = [&](int l, int j)         { return (const float*)d_in[61 + l * 4 + j]; };

  // ---- workspace carve ----
  char* wsp = (char*)d_ws;
  auto carve = [&](size_t bytes) -> void* {
    void* r = (void*)wsp; wsp += (bytes + 255) & ~(size_t)255; return r;
  };
  float* coordsA = (float*)carve((size_t)B * N0 * 3 * sizeof(float));
  float* coordsB = (float*)carve((size_t)B * 512 * 3 * sizeof(float));
  float* featsA  = (float*)carve((size_t)4 * 1024 * 1024);          // max 4MB
  float* featsB  = (float*)carve((size_t)4 * 1024 * 1024);
  int*   fidx    = (int*)  carve((size_t)B * 512 * sizeof(int));
  int*   gidx    = (int*)  carve((size_t)B * 512 * 128 * sizeof(int));
  float* meanB   = (float*)carve(2048 * sizeof(float));
  float* varB    = (float*)carve(2048 * sizeof(float));
  float* partS   = (float*)carve((size_t)128 * 2048 * sizeof(float));
  float* partQ   = (float*)carve((size_t)128 * 2048 * sizeof(float));
  bf16*  WP      = (bf16*) carve((size_t)2048 * 2048 * sizeof(bf16));
  bf16*  bufA    = (bf16*) carve((size_t)524288 * 160 * sizeof(bf16));  // 168MB
  bf16*  bufA2   = (bf16*) carve((size_t)524288 * 256 * sizeof(bf16));  // 268MB
  float* bufY    = (float*)carve((size_t)524288 * 256 * sizeof(float)); // 537MB

  auto gemm = [&](const bf16* Ain, const float* bias, float* Yp, int M, int Nn, int Kk) {
    dim3 g((unsigned)(Nn / 32), (unsigned)((M + 127) / 128));
    gemm_bf16_wmma<<<g, 256, 0, stream>>>(Ain, WP, bias, Yp, M, Nn, Kk);
  };
  auto bnstats = [&](const float* Yp, int M, int Nn) {
    int RB = (M + 1023) / 1024; if (RB < 1) RB = 1; if (RB > 128) RB = 128;
    dim3 gs((unsigned)(Nn / 64), (unsigned)RB);
    bn_stats_partial<<<gs, 256, 0, stream>>>(Yp, M, Nn, RB, partS, partQ);
    bn_stats_final<<<(Nn + 255) / 256, 256, 0, stream>>>(partS, partQ, Nn, RB, M, meanB, varB);
  };

  // Runs 3 MLP layers (GEMM -> global BN -> ReLU); final layer output left in bufY.
  auto run_mlp3 = [&](int st, int M, int cin0, int Kp0) {
    const bf16* Ain = bufA;
    bf16* outs[3] = {bufA2, bufA, bufA2};
    int Kk = Kp0, cinl = cin0;
    for (int l = 0; l < 3; ++l) {
      int Nn = mlps[st][l];
      size_t wpn = (size_t)Nn * Kk;
      pack_w_blocked<<<(unsigned)((wpn + 255) / 256), 256, 0, stream>>>(saP(st, l, 0), WP, cinl, Nn, wpn);
      gemm(Ain, saP(st, l, 1), bufY, M, Nn, Kk);
      bnstats(bufY, M, Nn);
      size_t tot = (size_t)M * Nn;
      bn_relu_kernel<<<(unsigned)((tot + 255) / 256), 256, 0, stream>>>(
          bufY, outs[l], meanB, varB, saP(st, l, 2), saP(st, l, 3), tot, Nn);
      Ain = outs[l]; Kk = Nn; cinl = Nn;
    }
  };

  // ---- stage 0 input ----
  split_xyz<<<(B * N0 + 255) / 256, 256, 0, stream>>>((const float*)d_in[0], coordsA, featsA, B, N0);

  float* coordsCur = coordsA; float* coordsNxt = coordsB;
  float* featsCur  = featsA;  float* featsNxt  = featsB;
  int curN = N0, Cf = 3;

  // ---- SA stages 0..3 ----
  for (int st = 0; st < 4; ++st) {
    int S = npointA[st], K = nsampA[st];
    float r2 = radA[st] * radA[st];
    fps_kernel<<<B, 256, 0, stream>>>(coordsCur, fidx, curN, S);
    int bs = B * S;
    ball_group<<<(bs + 255) / 256, 256, 0, stream>>>(coordsCur, fidx, coordsNxt, gidx,
                                                     B, curN, S, K, r2);
    int cin = 3 + Cf;
    int Kp = (int)pad32(cin);
    size_t M = (size_t)bs * K;
    build_grouped<<<(unsigned)((M + 255) / 256), 256, 0, stream>>>(
        coordsCur, featsCur, gidx, coordsNxt, bufA, curN, S, K, Cf, Kp, M);
    run_mlp3(st, (int)M, cin, Kp);
    int D = mlps[st][2];
    max_over_k<<<(bs * D + 255) / 256, 256, 0, stream>>>(bufY, featsNxt, bs, K, D);
    { float* t = coordsCur; coordsCur = coordsNxt; coordsNxt = t; }
    { float* t = featsCur;  featsCur  = featsNxt;  featsNxt  = t; }
    curN = S; Cf = D;
  }

  // ---- SA stage 4: group_all ----
  {
    int cin = 3 + Cf;                 // 1027
    int Kp = (int)pad32(cin);         // 1056
    int M = B * curN;                 // 512
    build_group_all<<<(M + 255) / 256, 256, 0, stream>>>(coordsCur, featsCur, bufA, M, Cf, Kp);
    run_mlp3(4, M, cin, Kp);
    max_over_k<<<(B * 2048 + 255) / 256, 256, 0, stream>>>(bufY, featsNxt, B, curN, 2048);
  }

  // ---- FC head: 3 x (16x2048 @ 2048x2048) ----
  cvt_bf16<<<(B * 2048 + 255) / 256, 256, 0, stream>>>(featsNxt, bufA, B * 2048);
  const bf16* xcur = bufA;
  bf16* xping[2] = {bufA2, bufA};
  for (int l = 0; l < 3; ++l) {
    size_t wpn = (size_t)2048 * 2048;
    pack_w_blocked<<<(unsigned)((wpn + 255) / 256), 256, 0, stream>>>(fcP(l, 0), WP, 2048, 2048, wpn);
    float* Yp = (l == 2) ? (float*)d_out : bufY;
    gemm(xcur, fcP(l, 1), Yp, B, 2048, 2048);
    if (l < 2) {
      bnstats(bufY, B, 2048);
      size_t tot = (size_t)B * 2048;
      bn_relu_kernel<<<(unsigned)((tot + 255) / 256), 256, 0, stream>>>(
          bufY, xping[l], meanB, varB, fcP(l, 2), fcP(l, 3), tot, 2048);
      xcur = xping[l];
    }
  }
}
